// WindowAttention_4114578670393
// MI455X (gfx1250) — matrix operand order repaired and checked by reading, not compiled, not run on hardware
//
#include <hip/hip_runtime.h>
#include <math.h>

// ---------------------------------------------------------------------------
// CDNA5 WMMA helpers (wave32, v_wmma_f32_16x16x32_f16)
// Register layouts used by these helpers (contiguous k, not the ISA table order):
//   A (16x32 MxK f16): lane L holds row m=L&15; element e -> k = e + 16*(L>>4)
//   B (32x16 KxN f16): lane L holds col n=L&15; element e -> k = e + 16*(L>>4)
//   C/D (16x16 f32):   element r -> m = r + 8*(L>>4), n = L&15
// ---------------------------------------------------------------------------
typedef _Float16 v16h __attribute__((ext_vector_type(16)));
typedef float    v8f  __attribute__((ext_vector_type(8)));
typedef int      v4i  __attribute__((vector_size(4 * sizeof(int))));

#if defined(__gfx1250__) && __has_builtin(__builtin_amdgcn_global_load_async_to_lds_b128)
#define ASYNC_LDS 1
#else
#define ASYNC_LDS 0
#endif

__device__ __forceinline__ v8f wmma_f32_f16(v16h a, v16h b, v8f c) {
  return __builtin_amdgcn_wmma_f32_16x16x32_f16(false, a, false, b, (short)0, c,
                                                false, false);
}

#if ASYNC_LDS
// 16B global -> LDS async copy (ASYNCcnt tracked, bypasses VGPRs).
// Builtin takes (v4i addrspace(1)*, v4i addrspace(3)*, imm offset, imm cpol).
__device__ __forceinline__ void async_b128(const void* gsrc, void* ldst) {
  __builtin_amdgcn_global_load_async_to_lds_b128(
      (__attribute__((address_space(1))) v4i*)gsrc,
      (__attribute__((address_space(3))) v4i*)ldst, 0, 0);
}
__device__ __forceinline__ void wait_async0() {
#if __has_builtin(__builtin_amdgcn_s_wait_asynccnt)
  __builtin_amdgcn_s_wait_asynccnt(0);
#else
  asm volatile("s_wait_asynccnt 0x0" ::: "memory");
#endif
}
#endif

// A fragment from row-major MxK storage (stride in f16 elements)
__device__ __forceinline__ v16h frag_a(const _Float16* base, int stride, int lane) {
  const int m = lane & 15, off = (lane >> 4) << 4;
  v16h a;
#pragma unroll
  for (int e = 0; e < 16; ++e)
    a[e] = base[m * stride + (e + off)];
  return a;
}

// B fragment where column n of B is a contiguous row of `base` (B = rows^T)
__device__ __forceinline__ v16h frag_b_rows(const _Float16* base, int stride, int lane) {
  const int n = lane & 15, off = (lane >> 4) << 4;
  v16h b;
#pragma unroll
  for (int e = 0; e < 16; ++e)
    b[e] = base[n * stride + e + off];
  return b;
}

// B fragment from row-major KxN storage (column n strided)
__device__ __forceinline__ v16h frag_b_cols(const _Float16* base, int stride, int lane) {
  const int n = lane & 15, off = (lane >> 4) << 4;
  v16h b;
#pragma unroll
  for (int e = 0; e < 16; ++e)
    b[e] = base[(e + off) * stride + n];
  return b;
}

// ---------------------------------------------------------------------------
// Kernel 0: weight conversion f32 -> f16 (qkv_w 768x256, proj_w 256x256)
// ---------------------------------------------------------------------------
__global__ __launch_bounds__(256) void cvt_weights_kernel(
    const float* __restrict__ qkv_w, const float* __restrict__ proj_w,
    _Float16* __restrict__ Wq, _Float16* __restrict__ Wp) {
  int i = blockIdx.x * 256 + threadIdx.x;
  if (i < 768 * 256) Wq[i] = (_Float16)qkv_w[i];
  if (i < 256 * 256) Wp[i] = (_Float16)proj_w[i];
}

// ---------------------------------------------------------------------------
// Kernel 1: CPB MLP -> bias[8][64][64] (16*sigmoid), plus per-head logit scales
// (rel tables are recomputed on device; they are not kernel inputs)
// ---------------------------------------------------------------------------
__global__ __launch_bounds__(256) void cpb_kernel(
    const float* __restrict__ w1, const float* __restrict__ b1,
    const float* __restrict__ w2, const float* __restrict__ logit_scale,
    float* __restrict__ bias, float* __restrict__ ls) {
  __shared__ float tbl[225][8];
  const int tid = threadIdx.x;
  if (tid < 225) {
    int i = tid / 15, j = tid % 15;
    float c0 = (float)(i - 7) * (8.0f / 7.0f);
    float c1 = (float)(j - 7) * (8.0f / 7.0f);
    const float inv_l2_8 = 1.0f / 3.0f;  // 1/log2(8)
    float s0 = (c0 > 0.f) ? 1.f : ((c0 < 0.f) ? -1.f : 0.f);
    float s1 = (c1 > 0.f) ? 1.f : ((c1 < 0.f) ? -1.f : 0.f);
    c0 = s0 * log2f(fabsf(c0) + 1.0f) * inv_l2_8;
    c1 = s1 * log2f(fabsf(c1) + 1.0f) * inv_l2_8;
    float acc[8] = {0.f, 0.f, 0.f, 0.f, 0.f, 0.f, 0.f, 0.f};
    for (int d = 0; d < 256; ++d) {
      float h = fmaxf(0.0f, c0 * w1[2 * d] + c1 * w1[2 * d + 1] + b1[d]);
#pragma unroll
      for (int hh = 0; hh < 8; ++hh) acc[hh] += h * w2[hh * 256 + d];
    }
#pragma unroll
    for (int hh = 0; hh < 8; ++hh) tbl[tid][hh] = acc[hh];
  }
  if (tid < 8) ls[tid] = expf(fminf(logit_scale[tid], 4.60517018598809f));  // ln(100)
  __syncthreads();
  for (int idx = tid; idx < 8 * 64 * 64; idx += 256) {
    int h = idx >> 12, rem = idx & 4095, i = rem >> 6, j = rem & 63;
    int d0 = (i >> 3) - (j >> 3) + 7;
    int d1 = (i & 7) - (j & 7) + 7;
    float v = tbl[d0 * 15 + d1][h];
    bias[idx] = 16.0f / (1.0f + expf(-v));
  }
}

// ---------------------------------------------------------------------------
// Kernel 2: QKV projection (GEMM 64x768x256 per window) fused with window
// partition.  Output scattered into q/k/v buffers [win][head][tok][hd] f16.
// One block = one window; 8 waves share 192 16x16 output tiles.  Each wave's
// mt is constant (tile stride 8), so the compiler keeps all 8 A-fragments
// resident and streams only B from L2.
// ---------------------------------------------------------------------------
__global__ __launch_bounds__(256) void qkv_gemm_kernel(
    const float* __restrict__ x, const _Float16* __restrict__ Wq,
    const float* __restrict__ qkv_b, _Float16* __restrict__ qb,
    _Float16* __restrict__ kb, _Float16* __restrict__ vb) {
  __shared__ _Float16 Xs[64][264];  // 64 tokens x 256 ch, +8 pad (bank spread)
  const int win = blockIdx.x;
  const int b = win >> 8, wy = (win >> 4) & 15, wx = win & 15;
  const int tid = threadIdx.x;

  // stage + convert window input (f32 -> f16, so no async path here)
  for (int i = tid; i < 64 * 256; i += 256) {
    int t = i & 63, c = i >> 6;
    int h = wy * 8 + (t >> 3), w = wx * 8 + (t & 7);
    Xs[t][c] = (_Float16)x[(((size_t)b * 256 + c) * 128 + h) * 128 + w];
  }
  __syncthreads();

  const int wave = tid >> 5, lane = tid & 31;
  const int n = lane & 15, half = lane >> 4;
  for (int tile = wave; tile < 192; tile += 8) {
    const int mt = tile & 3, nt = tile >> 2;  // nt in [0,48)
    v8f acc = {};
#pragma unroll
    for (int kt = 0; kt < 8; ++kt) {
      v16h A = frag_a(&Xs[mt * 16][kt * 32], 264, lane);
      v16h B = frag_b_rows(Wq + (size_t)(nt * 16) * 256 + kt * 32, 256, lane);
      acc = wmma_f32_f16(A, B, acc);
    }
    const int part = nt >> 4;                 // 0=q 1=k 2=v (16 n-tiles each)
    const int c = (nt & 15) * 16 + n;         // channel within part
    const int head = c >> 5, hd = c & 31;
    const int o = part * 256 + c;
    const float bv = qkv_b[o];
    _Float16* dst = (part == 0) ? qb : ((part == 1) ? kb : vb);
#pragma unroll
    for (int r = 0; r < 8; ++r) {
      int t = mt * 16 + r + 8 * half;
      dst[(((size_t)win * 8 + head) * 64 + t) * 32 + hd] = (_Float16)(acc[r] + bv);
    }
  }
}

// ---------------------------------------------------------------------------
// Kernel 3: attention.  One wave per (window, head).  32KB LDS:
//   Q[64][40] f16 | K[64][40] f16 | V[64][40] f16 | S[64][68] f32
//   P[64][72] f16 reuses the Q/K region after scores are computed.
// Logit scale folded into q normalization.  16 WMMA (QK^T) + 16 WMMA (PV).
// q/k/v staged with async global->LDS b128 copies (ASYNCcnt) when available.
// ---------------------------------------------------------------------------
__global__ __launch_bounds__(32) void attn_kernel(
    const _Float16* __restrict__ qb, const _Float16* __restrict__ kb,
    const _Float16* __restrict__ vb, const float* __restrict__ bias,
    const float* __restrict__ ls, _Float16* __restrict__ ob) {
  __shared__ __align__(16) unsigned char smem[32768];
  _Float16* Q = (_Float16*)smem;        // 64*40
  _Float16* K = Q + 64 * 40;            // 64*40
  _Float16* V = K + 64 * 40;            // 64*40
  float* S = (float*)(V + 64 * 40);     // 64*68 f32
  _Float16* P = (_Float16*)smem;        // 64*72 f16 (overlaps Q,K only)

  const int wh = blockIdx.x;            // win*8 + head
  const int head = wh & 7;
  const int lane = threadIdx.x;
  const size_t base = (size_t)wh * 64 * 32;

  // stage q,k,v rows (64B per row each)
#if ASYNC_LDS
  for (int t = lane; t < 64; t += 32) {
#pragma unroll
    for (int z = 0; z < 4; ++z) {
      async_b128(qb + base + (size_t)t * 32 + z * 8, Q + t * 40 + z * 8);
      async_b128(kb + base + (size_t)t * 32 + z * 8, K + t * 40 + z * 8);
      async_b128(vb + base + (size_t)t * 32 + z * 8, V + t * 40 + z * 8);
    }
  }
  wait_async0();
  __syncthreads();
#else
  for (int t = lane; t < 64; t += 32) {
    const uint4* sq = (const uint4*)(qb + base + (size_t)t * 32);
    const uint4* sk = (const uint4*)(kb + base + (size_t)t * 32);
    const uint4* sv = (const uint4*)(vb + base + (size_t)t * 32);
    uint4* dq = (uint4*)(Q + t * 40);
    uint4* dk = (uint4*)(K + t * 40);
    uint4* dv = (uint4*)(V + t * 40);
#pragma unroll
    for (int z = 0; z < 4; ++z) { dq[z] = sq[z]; dk[z] = sk[z]; dv[z] = sv[z]; }
  }
  __syncthreads();
#endif

  // cosine normalization; fold exp(min(logit_scale,ln100)) into q
  const float lsh = ls[head];
  for (int t = lane; t < 64; t += 32) {
    float sq = 0.f, sk = 0.f;
#pragma unroll
    for (int j = 0; j < 32; ++j) {
      float a = (float)Q[t * 40 + j]; sq += a * a;
      float c = (float)K[t * 40 + j]; sk += c * c;
    }
    float qs = lsh / fmaxf(sqrtf(sq), 1e-12f);
    float ks = 1.0f / fmaxf(sqrtf(sk), 1e-12f);
#pragma unroll
    for (int j = 0; j < 32; ++j) {
      Q[t * 40 + j] = (_Float16)((float)Q[t * 40 + j] * qs);
      K[t * 40 + j] = (_Float16)((float)K[t * 40 + j] * ks);
    }
  }
  __syncthreads();

  const int n = lane & 15, half = lane >> 4;
  const float* biash = bias + (size_t)head * 4096;

  // S = (ls*qn) @ kn^T  + bias      (K dim = 32 -> single WMMA per tile)
#pragma unroll
  for (int mt = 0; mt < 4; ++mt) {
    v16h A = frag_a(Q + mt * 16 * 40, 40, lane);
#pragma unroll
    for (int nt = 0; nt < 4; ++nt) {
      v8f acc = {};
      v16h B = frag_b_rows(K + nt * 16 * 40, 40, lane);
      acc = wmma_f32_f16(A, B, acc);
#pragma unroll
      for (int r = 0; r < 8; ++r) {
        int t = mt * 16 + r + 8 * half, u = nt * 16 + n;
        S[t * 68 + u] = acc[r] + biash[t * 64 + u];
      }
    }
  }
  __syncthreads();

  // row softmax -> P (f16), overwriting the Q/K region; exp computed once
  for (int t = lane; t < 64; t += 32) {
    float mx = -1e30f;
#pragma unroll
    for (int u = 0; u < 64; ++u) mx = fmaxf(mx, S[t * 68 + u]);
    float den = 0.f;
#pragma unroll
    for (int u = 0; u < 64; ++u) {
      float e = expf(S[t * 68 + u] - mx);
      S[t * 68 + u] = e;
      den += e;
    }
    float inv = 1.0f / den;
#pragma unroll
    for (int u = 0; u < 64; ++u)
      P[t * 72 + u] = (_Float16)(S[t * 68 + u] * inv);
  }
  __syncthreads();

  // O = P @ V   (M=64, N=32, K=64)
#pragma unroll
  for (int mt = 0; mt < 4; ++mt) {
    v8f acc0 = {}, acc1 = {};
#pragma unroll
    for (int kt = 0; kt < 2; ++kt) {
      v16h A = frag_a(P + mt * 16 * 72 + kt * 32, 72, lane);
      acc0 = wmma_f32_f16(A, frag_b_cols(V + kt * 32 * 40, 40, lane), acc0);
      acc1 = wmma_f32_f16(A, frag_b_cols(V + kt * 32 * 40 + 16, 40, lane), acc1);
    }
#pragma unroll
    for (int r = 0; r < 8; ++r) {
      int t = mt * 16 + r + 8 * half;
      ob[base + (size_t)t * 32 + n] = (_Float16)acc0[r];
      ob[base + (size_t)t * 32 + 16 + n] = (_Float16)acc1[r];
    }
  }
}

// ---------------------------------------------------------------------------
// Kernel 4: depthwise 3x3 PE conv on v (zero 'SAME' pad) + attention output,
// fused window-reverse; writes pixel-major t_img[b][h][w][c] f16 for proj GEMM.
// ---------------------------------------------------------------------------
__global__ __launch_bounds__(256) void pe_add_kernel(
    const _Float16* __restrict__ vb, const _Float16* __restrict__ ob,
    const float* __restrict__ pe_w, const float* __restrict__ pe_b,
    _Float16* __restrict__ timg) {
  size_t idx = (size_t)blockIdx.x * 256 + threadIdx.x;  // [b][h][w][c]
  const int c = (int)(idx & 255);
  const size_t p = idx >> 8;
  const int w = (int)(p & 127), h = (int)((p >> 7) & 127), b = (int)(p >> 14);
  const int head = c >> 5, hd = c & 31;
  float acc = pe_b[c];
#pragma unroll
  for (int kh = 0; kh < 3; ++kh) {
    int hh = h + kh - 1;
    if (hh < 0 || hh > 127) continue;
#pragma unroll
    for (int kw = 0; kw < 3; ++kw) {
      int ww = w + kw - 1;
      if (ww < 0 || ww > 127) continue;
      int win = (b * 16 + (hh >> 3)) * 16 + (ww >> 3);
      int t = ((hh & 7) << 3) | (ww & 7);
      float vv = (float)vb[(((size_t)win * 8 + head) * 64 + t) * 32 + hd];
      acc += vv * pe_w[c * 9 + kh * 3 + kw];
    }
  }
  int win = (b * 16 + (h >> 3)) * 16 + (w >> 3);
  int t = ((h & 7) << 3) | (w & 7);
  float o = (float)ob[(((size_t)win * 8 + head) * 64 + t) * 32 + hd];
  timg[idx] = (_Float16)(o + acc);
}

// ---------------------------------------------------------------------------
// Kernel 5: output projection GEMM (64 pixels x 256 out x 256 in per block),
// f32 result into d_out [B][C][H][W].  A-tile staged via async b128 copies;
// final stores are non-temporal (write-once 134MB stream, keep L2 for t_img).
// ---------------------------------------------------------------------------
__global__ __launch_bounds__(256) void proj_gemm_kernel(
    const _Float16* __restrict__ timg, const _Float16* __restrict__ Wp,
    const float* __restrict__ proj_b, float* __restrict__ out) {
  __shared__ _Float16 Xs[64][264];
  const int g = blockIdx.x;
  const size_t p0 = (size_t)g * 64;  // 64 pixels, same image (16384 | 64)
  const int tid = threadIdx.x;
#if ASYNC_LDS
  for (int i = tid; i < 2048; i += 256) {     // 2048 x 16B chunks
    int t = i >> 4, c16 = (i & 15) << 4;
    async_b128(timg + (p0 + t) * 256 + c16, &Xs[t][c16]);
  }
  wait_async0();
  __syncthreads();
#else
  for (int i = tid; i < 64 * 256; i += 256) {
    int t = i >> 8, c = i & 255;
    Xs[t][c] = timg[(p0 + t) * 256 + c];
  }
  __syncthreads();
#endif

  const int wave = tid >> 5, lane = tid & 31;
  const int n = lane & 15, half = lane >> 4;
  const int b = (int)(p0 >> 14);
  for (int tile = wave; tile < 64; tile += 8) {
    int mt = tile & 3, nt = tile >> 2;  // nt in [0,16)
    v8f acc = {};
#pragma unroll
    for (int kt = 0; kt < 8; ++kt) {
      v16h A = frag_a(&Xs[mt * 16][kt * 32], 264, lane);
      v16h B = frag_b_rows(Wp + (size_t)(nt * 16) * 256 + kt * 32, 256, lane);
      acc = wmma_f32_f16(A, B, acc);
    }
    int o = nt * 16 + n;
    float bv = proj_b[o];
#pragma unroll
    for (int r = 0; r < 8; ++r) {
      int m = mt * 16 + r + 8 * half;
      size_t hw = (p0 + m) & 16383;
      __builtin_nontemporal_store(acc[r] + bv,
                                  &out[((size_t)b * 256 + o) * 16384 + hw]);
    }
  }
}

// ---------------------------------------------------------------------------
extern "C" void kernel_launch(void* const* d_in, const int* in_sizes, int n_in,
                              void* d_out, int out_size, void* d_ws, size_t ws_size,
                              hipStream_t stream) {
  (void)in_sizes; (void)n_in; (void)out_size; (void)ws_size;
  const float* x           = (const float*)d_in[0];
  const float* qkv_w       = (const float*)d_in[1];
  const float* qkv_b       = (const float*)d_in[2];
  const float* proj_w      = (const float*)d_in[3];
  const float* proj_b      = (const float*)d_in[4];
  const float* pe_w        = (const float*)d_in[5];
  const float* pe_b        = (const float*)d_in[6];
  const float* logit_scale = (const float*)d_in[7];
  const float* cpb_w1      = (const float*)d_in[8];
  const float* cpb_b1      = (const float*)d_in[9];
  const float* cpb_w2      = (const float*)d_in[10];
  float* out = (float*)d_out;

  char* ws = (char*)d_ws;
  size_t off = 0;
  auto alloc = [&](size_t bytes) -> void* {
    void* pp = (void*)(ws + off);
    off += (bytes + 255) & ~(size_t)255;
    return pp;
  };
  const size_t QKVN = (size_t)2048 * 8 * 64 * 32;  // 33,554,432 elements
  _Float16* Wq   = (_Float16*)alloc(768 * 256 * sizeof(_Float16));
  _Float16* Wp   = (_Float16*)alloc(256 * 256 * sizeof(_Float16));
  _Float16* qb   = (_Float16*)alloc(QKVN * sizeof(_Float16));
  _Float16* kb   = (_Float16*)alloc(QKVN * sizeof(_Float16));
  _Float16* vb   = (_Float16*)alloc(QKVN * sizeof(_Float16));
  _Float16* ob   = (_Float16*)alloc(QKVN * sizeof(_Float16));
  _Float16* timg = (_Float16*)alloc(QKVN * sizeof(_Float16));
  float* bias = (float*)alloc(8 * 64 * 64 * sizeof(float));
  float* ls   = (float*)alloc(8 * sizeof(float));

  cvt_weights_kernel<<<768, 256, 0, stream>>>(qkv_w, proj_w, Wq, Wp);
  cpb_kernel<<<1, 256, 0, stream>>>(cpb_w1, cpb_b1, cpb_w2, logit_scale, bias, ls);
  qkv_gemm_kernel<<<2048, 256, 0, stream>>>(x, Wq, qkv_b, qb, kb, vb);
  attn_kernel<<<16384, 32, 0, stream>>>(qb, kb, vb, bias, ls, ob);
  pe_add_kernel<<<131072, 256, 0, stream>>>(vb, ob, pe_w, pe_b, timg);
  proj_gemm_kernel<<<2048, 256, 0, stream>>>(timg, Wp, proj_b, out);
}
